// Model_13812614824123
// MI455X (gfx1250) — matrix operand order repaired and checked by reading, not compiled, not run on hardware
//
#include <hip/hip_runtime.h>

// ---------------------------------------------------------------------------
// 4-layer GraphConv (DGL norm='none'):  h' = act( segsum(h[src] -> dst) @ W + b )
// Layers: 22->1000 relu, 1000->1000 relu, 1000->1000 relu, 1000->20 (rev graph)
// GEMMs: v_wmma_f32_16x16x32_bf16, double-buffered LDS fed by
// global_load_async_to_lds_b128 (ASYNCcnt pipeline) when available.
// ---------------------------------------------------------------------------

#define N_NODES 10000
#define N_EDGES 80000
#define MTILES  79               // ceil(10000/128)
#define MPAD    (MTILES * 128)   // 10112

typedef __attribute__((ext_vector_type(8)))  __bf16 v8bf;
typedef __attribute__((ext_vector_type(16))) __bf16 v16bf;
typedef __attribute__((ext_vector_type(8)))  float  v8f;

#if defined(__AMDGCN__) && __has_builtin(__builtin_amdgcn_global_load_async_to_lds_b128)
#define ASYNC_LDS 1
#else
#define ASYNC_LDS 0
#endif

// Builtin signature (from hipcc diagnostic): param0 = int4 __device__* (AS1
// global source), param1 = int4 __shared__* (AS3 LDS dest), then imm offset,
// imm cpol. Pointee must be a 16B int vector, not void.
typedef int v4i_ __attribute__((vector_size(16)));
typedef __attribute__((address_space(1))) v4i_ as1_v4i;
typedef __attribute__((address_space(3))) v4i_ as3_v4i;

// 16-byte global -> LDS copy. Async path: per-lane DMA into LDS, tracked by
// ASYNCcnt, no VGPR data round-trip, no per-chunk wait.
static __device__ __forceinline__ void copy16_g2l(void* lds, const void* g) {
#if ASYNC_LDS
  __builtin_amdgcn_global_load_async_to_lds_b128(
      (as1_v4i*)(uintptr_t)g,
      (as3_v4i*)(unsigned)(uintptr_t)lds, 0, 0);
#else
  *(v8bf*)lds = *(const v8bf*)g;
#endif
}

static __device__ __forceinline__ void async_wait0() {
#if ASYNC_LDS
#if __has_builtin(__builtin_amdgcn_s_wait_asynccnt)
  __builtin_amdgcn_s_wait_asynccnt(0);
#else
  asm volatile("s_wait_asynccnt 0" ::: "memory");
#endif
#endif
}

static __device__ __forceinline__ unsigned short f32_to_bf16(float f) {
  unsigned u = __float_as_uint(f);
  u += 0x7FFFu + ((u >> 16) & 1u);   // round-to-nearest-even
  return (unsigned short)(u >> 16);
}

// ---------------- aggregation: agg[d[e]] += h[s[e]] -------------------------

__global__ void scatter_scalar_kernel(const float* __restrict__ h,
                                      const int* __restrict__ s,
                                      const int* __restrict__ d,
                                      float* __restrict__ agg, int F) {
  int t = blockIdx.x * blockDim.x + threadIdx.x;
  int total = N_EDGES * F;
  if (t >= total) return;
  int e = t / F;
  int f = t - e * F;
  atomicAdd(&agg[(size_t)d[e] * F + f], h[(size_t)s[e] * F + f]);
}

__global__ void scatter_vec4_kernel(const float* __restrict__ h,
                                    const int* __restrict__ s,
                                    const int* __restrict__ d,
                                    float* __restrict__ agg, int F) {
  int C = F >> 2;
  int t = blockIdx.x * blockDim.x + threadIdx.x;
  int total = N_EDGES * C;
  if (t >= total) return;
  int e = t / C;
  int c = t - e * C;
  const float4 v = *(const float4*)(h + (size_t)s[e] * F + (size_t)c * 4);
  float* p = agg + (size_t)d[e] * F + (size_t)c * 4;
  atomicAdd(p + 0, v.x);
  atomicAdd(p + 1, v.y);
  atomicAdd(p + 2, v.z);
  atomicAdd(p + 3, v.w);
}

// --------------- pack f32 activations -> bf16, zero-padded ------------------

__global__ void pack_act_kernel(const float* __restrict__ agg,
                                unsigned short* __restrict__ out,
                                int F, int Kpad) {
  int cpr = Kpad >> 3;                  // 8-element chunks per row
  int t = blockIdx.x * blockDim.x + threadIdx.x;
  int total = MPAD * cpr;
  if (t >= total) return;
  int m = t / cpr;
  int c = t - m * cpr;
  int k0 = c * 8;
  unsigned short tmp[8];
#pragma unroll
  for (int j = 0; j < 8; ++j) {
    int k = k0 + j;
    float v = (m < N_NODES && k < F) ? agg[(size_t)m * F + k] : 0.0f;
    tmp[j] = f32_to_bf16(v);
  }
  uint4 pk;
  pk.x = (unsigned)tmp[0] | ((unsigned)tmp[1] << 16);
  pk.y = (unsigned)tmp[2] | ((unsigned)tmp[3] << 16);
  pk.z = (unsigned)tmp[4] | ((unsigned)tmp[5] << 16);
  pk.w = (unsigned)tmp[6] | ((unsigned)tmp[7] << 16);
  *(uint4*)(out + (size_t)t * 8) = pk;
}

// --------------- pack weights -> bf16, TRANSPOSED to N-major ----------------
// Wt[n][k] = W[k][n], zero-padded: B-tile copies and B-fragment reads become
// contiguous in K (matches ISA B layout: lane = column, two 8-K runs at half*8 and 16+half*8).

__global__ void pack_wt_kernel(const float* __restrict__ W,
                               unsigned short* __restrict__ Wt,
                               int Fin, int Fout, int Kpad, int Npad) {
  int t = blockIdx.x * blockDim.x + threadIdx.x;
  int total = Npad * Kpad;
  if (t >= total) return;
  int n = t / Kpad;
  int k = t - n * Kpad;
  float v = (n < Fout && k < Fin) ? W[(size_t)k * Fout + n] : 0.0f;
  Wt[t] = f32_to_bf16(v);
}

// --------------------------- WMMA GEMM --------------------------------------
// Block tile 128(M) x 128(N), 256 threads = 8 waves (4M x 2N), each wave owns
// 32x64 = 2x4 wmma_f32_16x16x32_bf16 accumulators. Double-buffered LDS with
// async global->LDS staging overlapped against the WMMAs.

template <int KPAD>
__global__ __launch_bounds__(256) void gcn_gemm_wmma_kernel(
    const unsigned short* __restrict__ Ab,   // [MPAD x KPAD] bf16 bits
    const unsigned short* __restrict__ Wt,   // [Npad x KPAD] bf16 bits (N-major)
    const float* __restrict__ bias,          // [Fout]
    float* __restrict__ out,                 // [N_NODES x Fout]
    int Fout, int doRelu) {
  __shared__ __align__(16) __bf16 As[2][128][40];  // 128 rows x 32 K (+8 pad)
  __shared__ __align__(16) __bf16 Bs[2][128][40];  // 128 cols x 32 K

  const int tid   = threadIdx.x;
  const int lane  = tid & 31;
  const int wv    = tid >> 5;
  const int waveM = wv >> 1;       // 0..3
  const int waveN = wv & 1;        // 0..1
  const int half  = lane >> 4;     // lane group 0/1
  const int r     = lane & 15;
  const int mBase = blockIdx.y * 128;
  const int nBase = blockIdx.x * 128;
  constexpr int NK = KPAD / 32;

  // Per-thread staging chunks: 512 16B chunks per tile, 2 per thread.
  const int c0row = tid >> 2, c0cc = (tid & 3) * 8;
  const int c1row = (tid + 256) >> 2, c1cc = ((tid + 256) & 3) * 8;
  const unsigned short* gA0 = Ab + (size_t)(mBase + c0row) * KPAD + c0cc;
  const unsigned short* gA1 = Ab + (size_t)(mBase + c1row) * KPAD + c1cc;
  const unsigned short* gB0 = Wt + (size_t)(nBase + c0row) * KPAD + c0cc;
  const unsigned short* gB1 = Wt + (size_t)(nBase + c1row) * KPAD + c1cc;

  v8f acc[2][4];
  for (int a = 0; a < 2; ++a)
    for (int b = 0; b < 4; ++b)
#pragma unroll
      for (int i = 0; i < 8; ++i) acc[a][b][i] = 0.0f;

  // Stage tile 0 into buffer 0.
  copy16_g2l(&As[0][c0row][c0cc], gA0);
  copy16_g2l(&As[0][c1row][c1cc], gA1);
  copy16_g2l(&Bs[0][c0row][c0cc], gB0);
  copy16_g2l(&Bs[0][c1row][c1cc], gB1);
  async_wait0();
  __syncthreads();

  for (int k = 0; k < NK; ++k) {
    const int cur = k & 1;
    // Stage tile k+1 into the other buffer (async; no wait yet).
    if (k + 1 < NK) {
      const int nxt = cur ^ 1;
      const int ko = (k + 1) * 32;
      copy16_g2l(&As[nxt][c0row][c0cc], gA0 + ko);
      copy16_g2l(&As[nxt][c1row][c1cc], gA1 + ko);
      copy16_g2l(&Bs[nxt][c0row][c0cc], gB0 + ko);
      copy16_g2l(&Bs[nxt][c1row][c1cc], gB1 + ko);
    }

    // A fragments (16x32 bf16): lanes 0-15 row M=r hold K {0..7,16..23},
    // lanes 16-31 hold K {8..15,24..31} -> two 16B LDS reads per fragment.
    v16bf af[2];
#pragma unroll
    for (int mf = 0; mf < 2; ++mf) {
      const __bf16* p = &As[cur][waveM * 32 + mf * 16 + r][half * 8];
      v8bf lo = *(const v8bf*)p;
      v8bf hi = *(const v8bf*)(p + 16);
      af[mf] = __builtin_shufflevector(lo, hi, 0, 1, 2, 3, 4, 5, 6, 7, 8, 9,
                                       10, 11, 12, 13, 14, 15);
    }
    // B fragments (32x16 bf16): lane = column N=r, K {half*8..half*8+7, 16+half*8..16+half*8+7}.
#pragma unroll
    for (int nf = 0; nf < 4; ++nf) {
      const __bf16* p = &Bs[cur][waveN * 64 + nf * 16 + r][half * 8];
      v8bf lo = *(const v8bf*)p;
      v8bf hi = *(const v8bf*)(p + 16);
      v16bf bfrag = __builtin_shufflevector(lo, hi, 0, 1, 2, 3, 4, 5, 6, 7, 8,
                                            9, 10, 11, 12, 13, 14, 15);
#pragma unroll
      for (int mf = 0; mf < 2; ++mf) {
        acc[mf][nf] = __builtin_amdgcn_wmma_f32_16x16x32_bf16(
            false, af[mf], false, bfrag, (short)0, acc[mf][nf], false, false);
      }
    }

    // Fragment ds_loads complete before the wmma consumers (s_wait_dscnt),
    // so by this barrier all reads of buffer `cur` are done; tile k+1 is
    // resident after the asynccnt wait.
    async_wait0();
    __syncthreads();
  }

  // Epilogue: C/D layout — VGPR i: lanes 0-15 (M=i, N=lane),
  // lanes 16-31 (M=i+8, N=lane-16).
#pragma unroll
  for (int mf = 0; mf < 2; ++mf) {
#pragma unroll
    for (int nf = 0; nf < 4; ++nf) {
      int col = nBase + waveN * 64 + nf * 16 + r;
#pragma unroll
      for (int i = 0; i < 8; ++i) {
        int row = mBase + waveM * 32 + mf * 16 + half * 8 + i;
        if (row < N_NODES && col < Fout) {
          float v = acc[mf][nf][i] + bias[col];
          if (doRelu) v = fmaxf(v, 0.0f);
          out[(size_t)row * Fout + col] = v;
        }
      }
    }
  }
}

// ---------------------------------------------------------------------------

template <int KPAD>
static void run_layer(const float* hin, int Fin, const float* W,
                      const float* bias, int Fout, int Ntiles, const int* sIdx,
                      const int* dIdx, float* hout, int doRelu, float* agg,
                      unsigned short* aggb, unsigned short* Wt,
                      hipStream_t stream) {
  // 1) zero aggregate buffer
  (void)hipMemsetAsync(agg, 0, (size_t)N_NODES * Fin * sizeof(float), stream);
  // 2) scatter-add h[sIdx] into agg[dIdx]
  if ((Fin & 3) == 0) {
    int total = N_EDGES * (Fin >> 2);
    scatter_vec4_kernel<<<(total + 255) / 256, 256, 0, stream>>>(hin, sIdx,
                                                                 dIdx, agg, Fin);
  } else {
    int total = N_EDGES * Fin;
    scatter_scalar_kernel<<<(total + 255) / 256, 256, 0, stream>>>(
        hin, sIdx, dIdx, agg, Fin);
  }
  // 3) pack activations -> bf16 [MPAD x KPAD], zero-padded
  {
    int total = MPAD * (KPAD >> 3);
    pack_act_kernel<<<(total + 255) / 256, 256, 0, stream>>>(agg, aggb, Fin,
                                                             KPAD);
  }
  // 4) pack + transpose weights -> bf16 [Npad x KPAD]
  {
    int Npad = Ntiles * 128;
    int total = Npad * KPAD;
    pack_wt_kernel<<<(total + 255) / 256, 256, 0, stream>>>(W, Wt, Fin, Fout,
                                                            KPAD, Npad);
  }
  // 5) WMMA GEMM (+bias, +relu)
  gcn_gemm_wmma_kernel<KPAD><<<dim3(Ntiles, MTILES), 256, 0, stream>>>(
      aggb, Wt, bias, hout, Fout, doRelu);
}

extern "C" void kernel_launch(void* const* d_in, const int* in_sizes, int n_in,
                              void* d_out, int out_size, void* d_ws,
                              size_t ws_size, hipStream_t stream) {
  (void)in_sizes; (void)n_in; (void)out_size; (void)ws_size;

  const float* x   = (const float*)d_in[0];
  const int*   src = (const int*)d_in[1];
  const int*   dst = (const int*)d_in[2];
  const float* W1  = (const float*)d_in[3];
  const float* b1  = (const float*)d_in[4];
  const float* W2  = (const float*)d_in[5];
  const float* b2  = (const float*)d_in[6];
  const float* W3  = (const float*)d_in[7];
  const float* b3  = (const float*)d_in[8];
  const float* W4  = (const float*)d_in[9];
  const float* b4  = (const float*)d_in[10];

  // Workspace carve-up (all sizes multiples of 256B).
  char* ws = (char*)d_ws;
  const size_t AGG_B = (size_t)N_NODES * 1000 * sizeof(float);        // 40 MB
  const size_t H_B   = (size_t)N_NODES * 1000 * sizeof(float);        // 40 MB
  const size_t AB_B  = (size_t)MPAD * 1024 * sizeof(unsigned short);  // 20.7 MB
  float*          agg  = (float*)ws;          ws += AGG_B;
  float*          hA   = (float*)ws;          ws += H_B;
  float*          hB   = (float*)ws;          ws += H_B;
  unsigned short* aggb = (unsigned short*)ws; ws += AB_B;
  unsigned short* Wt   = (unsigned short*)ws; // 1024*1024*2 = 2 MB

  // Layer 1: 22 -> 1000, relu
  run_layer<32>(x, 22, W1, b1, 1000, 8, src, dst, hA, 1, agg, aggb, Wt, stream);
  // Layer 2: 1000 -> 1000, relu
  run_layer<1024>(hA, 1000, W2, b2, 1000, 8, src, dst, hB, 1, agg, aggb, Wt,
                  stream);
  // Layer 3: 1000 -> 1000, relu
  run_layer<1024>(hB, 1000, W3, b3, 1000, 8, src, dst, hA, 1, agg, aggb, Wt,
                  stream);
  // Layer 4: reversed graph (agg[src] += h[dst]), 1000 -> 20, no act
  run_layer<1024>(hA, 1000, W4, b4, 20, 1, dst, src, (float*)d_out, 0, agg,
                  aggb, Wt, stream);
}
